// MOF_Net_41240275976362
// MI455X (gfx1250) — matrix operand order repaired and checked by reading, not compiled, not run on hardware
//
#include <hip/hip_runtime.h>
#include <stdint.h>

typedef __attribute__((ext_vector_type(16))) __bf16 v16bf;
typedef __attribute__((ext_vector_type(8)))  float  v8f;

union Frag16 {
  v16bf    v;
  uint4    q[2];
  uint32_t d[8];
  uint16_t u[16];
};

__device__ __forceinline__ uint16_t f2bf(float f) {
  uint32_t u = __builtin_bit_cast(uint32_t, f);
  u += 0x7FFFu + ((u >> 16) & 1u);          // round-to-nearest-even
  return (uint16_t)(u >> 16);
}

#define DD      64
#define KDIM    128
#define HID     128
#define WAVES   4
#define TILE_M  16
#define BLOCK_ROWS (WAVES * TILE_M)   // 64 edge rows per block tile

// ---------------------------------------------------------------------------
// Prep kernels
// ---------------------------------------------------------------------------
__global__ void prep_weights(const float* __restrict__ W1, const float* __restrict__ W2,
                             uint16_t* __restrict__ W1T, uint16_t* __restrict__ W2T) {
  int i = blockIdx.x * blockDim.x + threadIdx.x;
  if (i < HID * KDIM) {            // W1T[n][k] = W1[k][n], bf16
    int n = i >> 7, k = i & 127;
    W1T[i] = f2bf(W1[k * HID + n]);
  }
  if (i < DD * KDIM) {             // W2T[n][k] = W2[k][n], bf16
    int n = i >> 7, k = i & 127;
    W2T[i] = f2bf(W2[k * DD + n]);
  }
}

__global__ void init_copy_f4(const float4* __restrict__ s, float4* __restrict__ d,
                             long long n4) {
  long long i = blockIdx.x * (long long)blockDim.x + threadIdx.x;
  long long stride = (long long)gridDim.x * blockDim.x;
  for (; i < n4; i += stride) d[i] = s[i];
}

__global__ void zero_f(float* __restrict__ p, int n) {
  int i = blockIdx.x * blockDim.x + threadIdx.x;
  if (i < n) p[i] = 0.0f;
}

// ---------------------------------------------------------------------------
// Fused gather -> MLP (WMMA bf16) -> scatter-add
//   row r: in = [ xsrc[srcIdx[r]] (64) , eattr[r] (64) ]
//   msg   = relu(in @ W1 + b1) @ W2 + b2
//   outAcc[ remap? remap[dstIdx[r]] : dstIdx[r] ] += msg * scale   (atomic)
// ---------------------------------------------------------------------------
__launch_bounds__(WAVES * 32, 1)
__global__ void mlp_edge_scatter(
    const float*     __restrict__ xsrc,
    const float*     __restrict__ eattr,
    const long long* __restrict__ srcIdx,
    const long long* __restrict__ dstIdx,
    const long long* __restrict__ remap,    // nullptr, or batch[] for fused pooling
    const uint16_t*  __restrict__ gW1T,     // [128][128] bf16, N-major
    const uint16_t*  __restrict__ gW2T,     // [64][128]  bf16, N-major
    const float*     __restrict__ b1,
    const float*     __restrict__ b2,
    float*           __restrict__ outAcc,
    float scale,
    long long nEdges)
{
  __shared__ __align__(16) uint16_t sW1[HID * KDIM];             // 32 KB
  __shared__ __align__(16) uint16_t sW2[DD * KDIM];              // 16 KB
  __shared__ __align__(16) uint16_t sA[WAVES][TILE_M * KDIM];    // 16 KB

  const int tid  = threadIdx.x;
  const int wave = tid >> 5;
  const int lane = tid & 31;
  const int nlo  = lane & 15;
  const int klo  = (lane >= 16) ? 8  : 0;   // unused; A and B fragments both use kbHi
  const int kbHi = (lane >= 16) ? 16 : 0;   // B-fragment K base
  const int mhi  = (lane >= 16) ? 8  : 0;   // C/D M base

  // ---- stage weights into LDS ----
  {
    const uint4* s1 = (const uint4*)gW1T;
    uint4*       d1 = (uint4*)sW1;
    for (int i = tid; i < HID * KDIM / 8; i += blockDim.x) d1[i] = s1[i];
    const uint4* s2 = (const uint4*)gW2T;
    uint4*       d2 = (uint4*)sW2;
    for (int i = tid; i < DD * KDIM / 8; i += blockDim.x) d2[i] = s2[i];
  }
  // per-lane bias registers (N = nt*16 + nlo)
  float b1r[8], b2r[4];
  #pragma unroll
  for (int nt = 0; nt < 8; ++nt) b1r[nt] = b1[nt * 16 + nlo];
  #pragma unroll
  for (int nt = 0; nt < 4; ++nt) b2r[nt] = b2[nt * 16 + nlo];
  __syncthreads();

  const uint16_t* aRow = &sA[wave][(lane & 15) * KDIM];
  const long long nTiles = (nEdges + BLOCK_ROWS - 1) / BLOCK_ROWS;

  for (long long tile = blockIdx.x; tile < nTiles; tile += gridDim.x) {
    const long long rowBase = tile * BLOCK_ROWS + (long long)wave * TILE_M;

    // ---- gather 16 rows x 128 feats -> bf16 LDS tile ----
    {
      const int rr   = lane & 15;
      const int half = lane >> 4;          // 0: gathered x, 1: edge attr
      const long long r = rowBase + rr;
      uint16_t* dstp = &sA[wave][rr * KDIM + half * DD];
      if (r < nEdges) {
        const float* p = half ? (eattr + r * DD) : (xsrc + srcIdx[r] * DD);
        const float4* p4 = (const float4*)p;
        #pragma unroll
        for (int j = 0; j < DD / 4; ++j) {
          float4 v = p4[j];
          ushort4 o;
          o.x = f2bf(v.x); o.y = f2bf(v.y); o.z = f2bf(v.z); o.w = f2bf(v.w);
          *(ushort4*)(dstp + j * 4) = o;
        }
      } else {
        ushort4 z; z.x = z.y = z.z = z.w = 0;
        #pragma unroll
        for (int j = 0; j < DD / 4; ++j) *(ushort4*)(dstp + j * 4) = z;
      }
    }
    __syncthreads();

    // ---- GEMM1: 16x128 @ 128x128 -> hidden ----
    v8f acc1[8];
    {
      v8f z = {0.f,0.f,0.f,0.f,0.f,0.f,0.f,0.f};
      #pragma unroll
      for (int nt = 0; nt < 8; ++nt) acc1[nt] = z;
    }
    #pragma unroll
    for (int kk = 0; kk < 4; ++kk) {
      Frag16 fa;
      #pragma unroll
      for (int p = 0; p < 8; ++p) {        // 16-bit A layout: 2 K-values / VGPR
        int K = kk * 32 + kbHi + p * 2;
        fa.d[p] = *(const uint32_t*)(aRow + K);
      }
      #pragma unroll
      for (int nt = 0; nt < 8; ++nt) {
        Frag16 fb;
        const uint4* bp = (const uint4*)&sW1[(nt * 16 + nlo) * KDIM + kk * 32 + kbHi];
        fb.q[0] = bp[0]; fb.q[1] = bp[1];
        acc1[nt] = __builtin_amdgcn_wmma_f32_16x16x32_bf16(
            false, fa.v, false, fb.v, (short)0, acc1[nt], false, false);
      }
    }
    __syncthreads();                       // all sA reads done before overwrite

    // ---- bias + ReLU -> hidden tile back to LDS (bf16) ----
    #pragma unroll
    for (int nt = 0; nt < 8; ++nt) {
      #pragma unroll
      for (int i = 0; i < 8; ++i) {
        float h = acc1[nt][i] + b1r[nt];
        h = h > 0.0f ? h : 0.0f;
        sA[wave][(i + mhi) * KDIM + nt * 16 + nlo] = f2bf(h);
      }
    }
    __syncthreads();

    // ---- GEMM2: 16x128 @ 128x64 -> msg ----
    v8f acc2[4];
    {
      v8f z = {0.f,0.f,0.f,0.f,0.f,0.f,0.f,0.f};
      #pragma unroll
      for (int nt = 0; nt < 4; ++nt) acc2[nt] = z;
    }
    #pragma unroll
    for (int kk = 0; kk < 4; ++kk) {
      Frag16 fa;
      #pragma unroll
      for (int p = 0; p < 8; ++p) {
        int K = kk * 32 + kbHi + p * 2;
        fa.d[p] = *(const uint32_t*)(aRow + K);
      }
      #pragma unroll
      for (int nt = 0; nt < 4; ++nt) {
        Frag16 fb;
        const uint4* bp = (const uint4*)&sW2[(nt * 16 + nlo) * KDIM + kk * 32 + kbHi];
        fb.q[0] = bp[0]; fb.q[1] = bp[1];
        acc2[nt] = __builtin_amdgcn_wmma_f32_16x16x32_bf16(
            false, fa.v, false, fb.v, (short)0, acc2[nt], false, false);
      }
    }

    // ---- scatter-add (segment_sum via f32 atomics) ----
    #pragma unroll
    for (int i = 0; i < 8; ++i) {
      const long long r = rowBase + i + mhi;
      if (r < nEdges) {
        long long orow = dstIdx[r];
        if (remap) orow = remap[orow];
        float* op = outAcc + orow * DD + nlo;
        #pragma unroll
        for (int nt = 0; nt < 4; ++nt)
          atomicAdd(op + nt * 16, (acc2[nt][i] + b2r[nt]) * scale);
      }
    }
    __syncthreads();                       // guard sA vs next tile's gather
  }
}

// ---------------------------------------------------------------------------
extern "C" void kernel_launch(void* const* d_in, const int* in_sizes, int n_in,
                              void* d_out, int out_size, void* d_ws, size_t ws_size,
                              hipStream_t stream) {
  const float*     x1    = (const float*)d_in[0];
  const float*     x2    = (const float*)d_in[1];
  const long long* ei1   = (const long long*)d_in[2];   // [2, E1]
  const long long* ei2   = (const long long*)d_in[3];   // [2, E2]
  const long long* batch = (const long long*)d_in[4];   // [N1]
  const float*     ea1   = (const float*)d_in[5];
  const float*     ea2   = (const float*)d_in[6];
  const float*     W1    = (const float*)d_in[7];
  const float*     b1    = (const float*)d_in[8];
  const float*     W2    = (const float*)d_in[9];
  const float*     b2    = (const float*)d_in[10];

  const long long E1 = in_sizes[5] / DD;
  const long long E2 = in_sizes[6] / DD;

  // workspace: e1 accumulator [E1, 64] f32, then bf16 transposed weights
  float* e1 = (float*)d_ws;
  size_t e1Bytes = ((size_t)E1 * DD * sizeof(float) + 255) & ~(size_t)255;
  uint16_t* W1T = (uint16_t*)((char*)d_ws + e1Bytes);
  uint16_t* W2T = W1T + HID * KDIM;

  // prep: bf16 weights, e1 = edge_attr1, out = 0
  prep_weights<<<(HID * KDIM + 255) / 256, 256, 0, stream>>>(W1, W2, W1T, W2T);
  long long n4 = E1 * DD / 4;
  init_copy_f4<<<2048, 256, 0, stream>>>((const float4*)ea1, (float4*)e1, n4);
  zero_f<<<(out_size + 255) / 256, 256, 0, stream>>>((float*)d_out, out_size);

  // stage 1: line-graph conv on angles -> accumulate into bond features e1
  long long t2 = (E2 + BLOCK_ROWS - 1) / BLOCK_ROWS;
  int g1 = (int)(t2 < 4096 ? t2 : 4096);
  mlp_edge_scatter<<<g1, WAVES * 32, 0, stream>>>(
      x2, ea2, ei2 /*src2*/, ei2 + E2 /*dst2*/, nullptr,
      W1T, W2T, b1, b2, e1, 1.0f, E2);

  // stage 2: node conv on bonds, fused with global-add-pool (/2)
  long long t1 = (E1 + BLOCK_ROWS - 1) / BLOCK_ROWS;
  int g2 = (int)(t1 < 4096 ? t1 : 4096);
  mlp_edge_scatter<<<g2, WAVES * 32, 0, stream>>>(
      x1, e1, ei1 /*src1*/, ei1 + E1 /*dst1*/, batch,
      W1T, W2T, b1, b2, (float*)d_out, 0.5f, E1);
}
